// TAdaBlock_10917806866929
// MI455X (gfx1250) — matrix operand order repaired and checked by reading, not compiled, not run on hardware
//
#include <hip/hip_runtime.h>

// ---------------------------------------------------------------------------
// Types for CDNA5 WMMA (wave32, 16x16x32 bf16 -> f32)
// ---------------------------------------------------------------------------
typedef __bf16 bf16_t;
typedef __attribute__((ext_vector_type(8)))  __bf16 bf16x8;
typedef __attribute__((ext_vector_type(16))) __bf16 bf16x16;
typedef __attribute__((ext_vector_type(8)))  float  f32x8;

#define L_TOK 577
#define BT_   128
#define D_    768
#define B_    16
#define T_    8
#define HW_   576
#define CM_   192
#define CR_   48
#define MTOT  (L_TOK * BT_)   // 73856 rows = 128 * 577

// ---------------------------------------------------------------------------
// Async global->LDS copy (CDNA5 GLOBAL_LOAD_ASYNC_TO_LDS_B128, ASYNCcnt).
// Probe round 2 revealed the builtin's signature: arg0 is a NON-const
// pointer to int-vector(16B) in the global address space (printed by clang
// as "int __vector(4) __device__ *"), arg1 the LDS-side pointer, then two
// immediate ints (offset, cpol).
// ---------------------------------------------------------------------------
#if __has_builtin(__builtin_amdgcn_global_load_async_to_lds_b128) && \
    __has_builtin(__builtin_amdgcn_s_wait_asynccnt)
#define HAVE_ASYNC_LDS 1
typedef int v4i_vs __attribute__((vector_size(16)));
typedef __attribute__((address_space(1))) v4i_vs g_v4i;
typedef __attribute__((address_space(3))) v4i_vs l_v4i;
static __device__ __forceinline__ void async_copy_b128(const void* g, void* l) {
    __builtin_amdgcn_global_load_async_to_lds_b128(
        (g_v4i*)(void*)(size_t)(const char*)g, (l_v4i*)l, 0, 0);
}
static __device__ __forceinline__ void async_wait0() {
    __builtin_amdgcn_s_wait_asynccnt(0);
}
#else
#define HAVE_ASYNC_LDS 0
static __device__ __forceinline__ void async_copy_b128(const void* g, void* l) {
    *(bf16x8*)l = *(const bf16x8*)g;    // fallback: VGPR round-trip
}
static __device__ __forceinline__ void async_wait0() {}
#endif

// ---------------------------------------------------------------------------
// Weight conversion + transpose: src f32 [K][N] -> dst bf16 [N][K].
// Makes every 32-K LDS tile row a contiguous 64B global run (async-friendly)
// ---------------------------------------------------------------------------
__global__ void cvt_transpose_bf16(const float* __restrict__ src,
                                   unsigned short* __restrict__ dst,
                                   int K, int N) {
    int i = blockIdx.x * blockDim.x + threadIdx.x;   // over N*K
    if (i < N * K) {
        int n = i / K, k = i % K;
        bf16_t v = (bf16_t)src[(size_t)k * N + n];
        dst[i] = __builtin_bit_cast(unsigned short, v);
    }
}

// ---------------------------------------------------------------------------
// GEMM: C[M,N] = A[M,K] * Bt[N,K](bf16) + bias[N]
// Block tile 128(M) x 192(N), K-step 32; 256 threads = 8 wave32s (4x2 grid);
// each wave owns 32x96 = 2 A-frags x 6 B-frags = 12 WMMA accumulators.
// A is f32 (converted on the fly) or bf16 (async-copied) per template arg.
// ---------------------------------------------------------------------------
template <bool A_BF16>
__global__ __launch_bounds__(256)
void gemm_wmma(const void* __restrict__ Ain,
               const unsigned short* __restrict__ Bt_u,
               const float* __restrict__ bias,
               float* __restrict__ C,
               int M, int N, int K) {
    const bf16_t* Bt = (const bf16_t*)Bt_u;
    __shared__ bf16_t sA[128 * 40];   // 10.0 KB, [m][k], stride-padded
    __shared__ bf16_t sB[192 * 40];   // 15.0 KB, [n][k] (B pre-transposed)

    const int nBN  = N / 192;
    const int bm   = blockIdx.x / nBN;
    const int bn   = blockIdx.x % nBN;
    const int tid  = threadIdx.x;
    const int lane = tid & 31;
    const int wave = tid >> 5;
    const int waveM = wave >> 1;       // 0..3 -> 32-row slab
    const int waveN = wave & 1;        // 0..1 -> 96-col slab

    f32x8 acc[12] = {};                // [mi*6 + s]

    // f32-A staging: 2 threads per row, 16 floats each
    const int rowAf = tid >> 1;            // 0..127
    const int kcAf  = (tid & 1) << 4;      // 0,16
    const float* gAf = (const float*)Ain + (size_t)(bm * 128 + rowAf) * K + kcAf;

    for (int k0 = 0; k0 < K; k0 += 32) {
        // ---- stage A tile: 128x32 ----
        if (A_BF16) {
            const unsigned short* Ab = (const unsigned short*)Ain;
            #pragma unroll
            for (int v = 0; v < 2; ++v) {               // 512 16B chunks / 256 thr
                const int ch  = tid + (v << 8);
                const int row = ch >> 2;
                const int kc8 = (ch & 3) << 3;
                async_copy_b128(Ab + (size_t)(bm * 128 + row) * K + k0 + kc8,
                                &sA[row * 40 + kc8]);
            }
        } else {
            float4 a0 = *(const float4*)(gAf + k0);
            float4 a1 = *(const float4*)(gAf + k0 + 4);
            float4 a2 = *(const float4*)(gAf + k0 + 8);
            float4 a3 = *(const float4*)(gAf + k0 + 12);
            bf16x8 v0, v1;
            v0[0] = (bf16_t)a0.x; v0[1] = (bf16_t)a0.y;
            v0[2] = (bf16_t)a0.z; v0[3] = (bf16_t)a0.w;
            v0[4] = (bf16_t)a1.x; v0[5] = (bf16_t)a1.y;
            v0[6] = (bf16_t)a1.z; v0[7] = (bf16_t)a1.w;
            v1[0] = (bf16_t)a2.x; v1[1] = (bf16_t)a2.y;
            v1[2] = (bf16_t)a2.z; v1[3] = (bf16_t)a2.w;
            v1[4] = (bf16_t)a3.x; v1[5] = (bf16_t)a3.y;
            v1[6] = (bf16_t)a3.z; v1[7] = (bf16_t)a3.w;
            *(bf16x8*)&sA[rowAf * 40 + kcAf]     = v0;
            *(bf16x8*)&sA[rowAf * 40 + kcAf + 8] = v1;
            if (k0 + 32 < K) __builtin_prefetch(gAf + k0 + 32, 0, 1);
        }

        // ---- stage B tile: 192 rows x 32 K (contiguous 64B rows in global) ----
        #pragma unroll
        for (int v = 0; v < 3; ++v) {                   // 768 16B chunks / 256 thr
            const int ch  = tid + (v << 8);
            const int n   = ch >> 2;
            const int kc8 = (ch & 3) << 3;
            async_copy_b128(Bt + (size_t)n * K + (size_t)bn * 192 * K + k0 + kc8,
                            &sB[n * 40 + kc8]);
        }

        async_wait0();
        __syncthreads();

        // ---- A fragments: lane l = row (l&15), K halves split by (l>>4) ----
        bf16x16 afrag[2];
        #pragma unroll
        for (int mi = 0; mi < 2; ++mi) {
            const int m    = waveM * 32 + mi * 16 + (lane & 15);
            const int koff = (lane >> 4) << 3;          // 0 or 8
            bf16x8 lo = *(const bf16x8*)&sA[m * 40 + koff];
            bf16x8 hi = *(const bf16x8*)&sA[m * 40 + 16 + koff];
            #pragma unroll
            for (int i = 0; i < 8; ++i) { afrag[mi][i] = lo[i]; afrag[mi][8 + i] = hi[i]; }
        }

        // ---- 12 WMMAs: each B fragment reused for both A fragments ----
        #pragma unroll
        for (int s = 0; s < 6; ++s) {
            const int n     = waveN * 96 + s * 16 + (lane & 15);
            const int koff2 = (lane >> 4) << 3;         // 0 or 8
            bf16x8 lo = *(const bf16x8*)&sB[n * 40 + koff2];
            bf16x8 hi = *(const bf16x8*)&sB[n * 40 + 16 + koff2];
            bf16x16 bfrag;
            #pragma unroll
            for (int i = 0; i < 8; ++i) { bfrag[i] = lo[i]; bfrag[8 + i] = hi[i]; }
            #pragma unroll
            for (int mi = 0; mi < 2; ++mi)
                acc[mi * 6 + s] = __builtin_amdgcn_wmma_f32_16x16x32_bf16(
                    false, afrag[mi], false, bfrag, (short)0, acc[mi * 6 + s],
                    false, false);
        }
        __syncthreads();
    }

    // ---- epilogue ----
    #pragma unroll
    for (int mi = 0; mi < 2; ++mi) {
        const int mBase = bm * 128 + waveM * 32 + mi * 16 + ((lane >> 4) << 3);
        #pragma unroll
        for (int s = 0; s < 6; ++s) {
            const int n  = bn * 192 + waveN * 96 + s * 16 + (lane & 15);
            const float bv = bias[n];
            #pragma unroll
            for (int r = 0; r < 8; ++r)
                C[(size_t)(mBase + r) * N + n] = acc[mi * 6 + s][r] + bv;
        }
    }
}

// ---------------------------------------------------------------------------
// d[bt][c] = mean over 576 spatial positions of h[1+p][bt][c]
// ---------------------------------------------------------------------------
__global__ void reduce_d_kernel(const float* __restrict__ h, float* __restrict__ d) {
    const int bt = blockIdx.x;     // 0..127
    const int c  = threadIdx.x;    // 0..191
    const float* p = h + (size_t)(BT_ + bt) * CM_ + c;
    float s = 0.f;
    for (int sp = 0; sp < HW_; ++sp) s += p[(size_t)sp * BT_ * CM_];
    d[bt * CM_ + c] = s * (1.0f / (float)HW_);
}

// ---------------------------------------------------------------------------
// Tiny temporal-adaptive path: g = mean_t(d) @ w_g + b_g;
// a = relu(conv1d(d+g, w_a) + b_a); alpha = conv1d(a, w_b) + 1.
// ---------------------------------------------------------------------------
__global__ void modulator_kernel(const float* __restrict__ d,
                                 const float* __restrict__ w_g, const float* __restrict__ b_g,
                                 const float* __restrict__ w_a, const float* __restrict__ b_a,
                                 const float* __restrict__ w_b,
                                 float* __restrict__ alpha) {
    __shared__ float sd[T_][CM_];
    __shared__ float sdm[CM_];
    __shared__ float sg[CM_];
    __shared__ float sa[T_][CR_];
    const int b = blockIdx.x;    // 0..15
    const int c = threadIdx.x;   // 0..191

    for (int t = 0; t < T_; ++t) sd[t][c] = d[(b * T_ + t) * CM_ + c];
    __syncthreads();

    float m = 0.f;
    #pragma unroll
    for (int t = 0; t < T_; ++t) m += sd[t][c];
    sdm[c] = m * (1.0f / (float)T_);
    __syncthreads();

    float gacc = b_g[c];
    for (int ci = 0; ci < CM_; ++ci) gacc += sdm[ci] * w_g[ci * CM_ + c];
    sg[c] = gacc;
    __syncthreads();

    for (int o = c; o < T_ * CR_; o += CM_) {
        const int t  = o / CR_;
        const int co = o % CR_;
        float acc = b_a[co];
        #pragma unroll
        for (int dk = 0; dk < 3; ++dk) {
            const int ts = t + dk - 1;
            if (ts < 0 || ts >= T_) continue;
            const float* wa = w_a + dk * CM_ * CR_ + co;
            for (int ci = 0; ci < CM_; ++ci)
                acc += (sd[ts][ci] + sg[ci]) * wa[ci * CR_];
        }
        sa[t][co] = fmaxf(acc, 0.0f);
    }
    __syncthreads();

    for (int t = 0; t < T_; ++t) {
        float acc = 1.0f;
        #pragma unroll
        for (int dk = 0; dk < 3; ++dk) {
            const int ts = t + dk - 1;
            if (ts < 0 || ts >= T_) continue;
            const float* wb = w_b + dk * CR_ * CM_ + c;
            for (int ci = 0; ci < CR_; ++ci)
                acc += sa[ts][ci] * wb[ci * CM_];
        }
        alpha[(b * T_ + t) * CM_ + c] = acc;
    }
}

// ---------------------------------------------------------------------------
// Depthwise 3x3 (SAME) * alpha + b_conv, concat cls row, SiLU(1.702).
// Writes activation as bf16 so GEMM2's A operand is async-LDS-copyable.
// ---------------------------------------------------------------------------
__global__ void dw_silu_kernel(const float* __restrict__ h,
                               const float* __restrict__ w_base,
                               const float* __restrict__ b_conv,
                               const float* __restrict__ alpha,
                               unsigned short* __restrict__ act) {
    const int bid = blockIdx.x;     // 0..73855
    const int p   = bid >> 7;       // token row 0..576
    const int bt  = bid & 127;
    const int c   = threadIdx.x;
    const size_t rowstride = (size_t)BT_ * CM_;

    float v;
    if (p == 0) {
        v = h[(size_t)bt * CM_ + c];   // cls token passthrough
    } else {
        const int sp = p - 1, hh = sp / 24, ww = sp % 24;
        float y = 0.f;
        #pragma unroll
        for (int i = 0; i < 3; ++i) {
            const int h2 = hh + i - 1;
            if (h2 < 0 || h2 >= 24) continue;
            #pragma unroll
            for (int j = 0; j < 3; ++j) {
                const int w2 = ww + j - 1;
                if (w2 < 0 || w2 >= 24) continue;
                y += h[(size_t)(1 + h2 * 24 + w2) * rowstride + (size_t)bt * CM_ + c]
                     * w_base[(i * 3 + j) * CM_ + c];
            }
        }
        v = y * alpha[bt * CM_ + c] + b_conv[c];
    }
    const float s = v / (1.0f + __expf(-1.702f * v));   // x * sigmoid(1.702 x)
    bf16_t sb = (bf16_t)s;
    act[(size_t)(p * BT_ + bt) * CM_ + c] = __builtin_bit_cast(unsigned short, sb);
}

// ---------------------------------------------------------------------------
// Host-side orchestration
// ---------------------------------------------------------------------------
extern "C" void kernel_launch(void* const* d_in, const int* in_sizes, int n_in,
                              void* d_out, int out_size, void* d_ws, size_t ws_size,
                              hipStream_t stream) {
    (void)in_sizes; (void)n_in; (void)out_size; (void)ws_size;
    const float* x      = (const float*)d_in[0];
    const float* w_pre  = (const float*)d_in[1];
    const float* b_pre  = (const float*)d_in[2];
    const float* w_base = (const float*)d_in[3];
    const float* b_conv = (const float*)d_in[4];
    const float* w_g    = (const float*)d_in[5];
    const float* b_g    = (const float*)d_in[6];
    const float* w_a    = (const float*)d_in[7];
    const float* b_a    = (const float*)d_in[8];
    const float* w_b    = (const float*)d_in[9];
    const float* w_proj = (const float*)d_in[10];
    const float* b_proj = (const float*)d_in[11];
    float* out = (float*)d_out;

    char* ws = (char*)d_ws;
    auto alignup = [](size_t v) { return (v + 255) & ~(size_t)255; };
    size_t off = 0;
    float*          h        = (float*)(ws + off);          off += alignup((size_t)MTOT * CM_ * 4);
    unsigned short* act      = (unsigned short*)(ws + off); off += alignup((size_t)MTOT * CM_ * 2);
    unsigned short* wpre_bt  = (unsigned short*)(ws + off); off += alignup((size_t)D_ * CM_ * 2);
    unsigned short* wproj_bt = (unsigned short*)(ws + off); off += alignup((size_t)CM_ * D_ * 2);
    float*          dbuf     = (float*)(ws + off);          off += alignup((size_t)B_ * T_ * CM_ * 4);
    float*          alpha    = (float*)(ws + off);          off += alignup((size_t)BT_ * CM_ * 4);

    // 1) weights -> bf16, transposed to [N][K]
    const int nW = D_ * CM_;  // 147456 each
    cvt_transpose_bf16<<<(nW + 255) / 256, 256, 0, stream>>>(w_pre,  wpre_bt,  D_,  CM_);
    cvt_transpose_bf16<<<(nW + 255) / 256, 256, 0, stream>>>(w_proj, wproj_bt, CM_, D_);

    // 2) h = x @ w_pre + b_pre        (73856 x 768) * (768 x 192)
    gemm_wmma<false><<<(MTOT / 128) * (CM_ / 192), 256, 0, stream>>>(
        x, wpre_bt, b_pre, h, MTOT, CM_, D_);

    // 3) spatial mean -> d
    reduce_d_kernel<<<BT_, CM_, 0, stream>>>(h, dbuf);

    // 4) tiny adaptive path -> alpha
    modulator_kernel<<<B_, CM_, 0, stream>>>(dbuf, w_g, b_g, w_a, b_a, w_b, alpha);

    // 5) depthwise conv * alpha + bias, concat cls, SiLU -> act (bf16)
    dw_silu_kernel<<<L_TOK * BT_, CM_, 0, stream>>>(h, w_base, b_conv, alpha, act);

    // 6) out = act @ w_proj + b_proj  (73856 x 192) * (192 x 768)
    gemm_wmma<true><<<(MTOT / 128) * (D_ / 192), 256, 0, stream>>>(
        act, wproj_bt, b_proj, out, MTOT, D_, CM_);
}
